// GraniteMoeMoE_11201274708062
// MI455X (gfx1250) — matrix operand order repaired and checked by reading, not compiled, not run on hardware
//
#include <hip/hip_runtime.h>
#include <math.h>

// Problem constants (GraniteMoE / JetMoE MoE layer)
#define T_TOK 4096   // B*S tokens
#define DIM   1024   // hidden D
#define HID   512    // intermediate H
#define TWO_H 1024   // 2H
#define NEXP  8
#define TOPK  2

#define KSTEP  64    // K elements staged per pipeline stage
#define LDSROW 72    // bf16 units per LDS row (64 data + 8 pad -> 144B stride)

typedef __attribute__((ext_vector_type(16))) __bf16 v16bf;
typedef __attribute__((ext_vector_type(8)))  float  v8f;

union Frag {
    v16bf v;
    unsigned int u[8];
};

// fp32 -> bf16 round-to-nearest-even (used only in one-shot convert + epilogue)
__device__ __forceinline__ unsigned int f2bf(float f) {
    unsigned int u = __float_as_uint(f);
    return (u + 0x7FFFu + ((u >> 16) & 1u)) >> 16;
}
__device__ __forceinline__ unsigned int pack2(float lo, float hi) {
    return f2bf(lo) | (f2bf(hi) << 16);
}

// ---- gfx1250 async global->LDS copy (GLOBAL_LOAD_ASYNC_TO_LDS_B128) --------
#ifndef USE_ASYNC_LDS
#if defined(__has_builtin)
#if __has_builtin(__builtin_amdgcn_global_load_async_to_lds_b128) && \
    __has_builtin(__builtin_amdgcn_s_wait_asynccnt)
#define USE_ASYNC_LDS 1
#endif
#endif
#endif
#ifndef USE_ASYNC_LDS
#define USE_ASYNC_LDS 0
#endif

// Builtin parameter type per hipcc diagnostic: 'int __attribute__((vector_size(16))) *'
typedef int v4i __attribute__((vector_size(16)));
typedef __attribute__((address_space(1))) v4i gv4i;  // global
typedef __attribute__((address_space(3))) v4i lv4i;  // LDS

__device__ __forceinline__ gv4i* to_gbl4(const void* p) {
    return (gv4i*)(unsigned long long)(size_t)p;
}
// generic pointer to __shared__ is {aperture_hi, lds_offset}; low 32 bits are
// the LDS byte offset == the 32-bit AS(3) pointer value.
__device__ __forceinline__ lv4i* to_lds4(void* p) {
    return (lv4i*)(unsigned int)(size_t)p;
}

__device__ __forceinline__ void stage16(const unsigned short* src,
                                        unsigned short* dst) {
#if USE_ASYNC_LDS
    __builtin_amdgcn_global_load_async_to_lds_b128(to_gbl4(src), to_lds4(dst), 0, 0);
#else
    *(uint4*)dst = *(const uint4*)src;
#endif
}
__device__ __forceinline__ void wait_async() {
#if USE_ASYNC_LDS
    __builtin_amdgcn_s_wait_asynccnt(0);
#endif
}

// ---------------------------------------------------------------------------
// One-shot fp32 -> bf16 conversion (x, w_in, w_out). 8 elems per thread.
// ---------------------------------------------------------------------------
__global__ __launch_bounds__(256) void cvt_bf16_kernel(
    const float* __restrict__ src, unsigned short* __restrict__ dst, int n8) {
    int i = blockIdx.x * 256 + threadIdx.x;
    if (i >= n8) return;
    const float4* s = (const float4*)src + (size_t)i * 2;
    float4 f0 = s[0], f1 = s[1];
    uint4 o;
    o.x = pack2(f0.x, f0.y);
    o.y = pack2(f0.z, f0.w);
    o.z = pack2(f1.x, f1.y);
    o.w = pack2(f1.z, f1.w);
    ((uint4*)dst)[i] = o;
}

// ---------------------------------------------------------------------------
// Router: one wave32 per token. logits = x . w_router^T, top-2, softmax gates.
// ---------------------------------------------------------------------------
__global__ __launch_bounds__(256) void router_kernel(
    const float* __restrict__ x, const float* __restrict__ wr,
    float* __restrict__ logits_out, int* __restrict__ topi,
    float* __restrict__ tgate) {
    const int wid  = threadIdx.x >> 5;
    const int lane = threadIdx.x & 31;
    const int t    = blockIdx.x * 8 + wid;
    const float* xt = x + (size_t)t * DIM;

    float acc[NEXP];
#pragma unroll
    for (int e = 0; e < NEXP; ++e) acc[e] = 0.0f;

    for (int d = lane * 4; d < DIM; d += 128) {
        float4 xv = *(const float4*)(xt + d);
#pragma unroll
        for (int e = 0; e < NEXP; ++e) {
            float4 wv = *(const float4*)(wr + e * DIM + d);
            acc[e] += xv.x * wv.x + xv.y * wv.y + xv.z * wv.z + xv.w * wv.w;
        }
    }
#pragma unroll
    for (int e = 0; e < NEXP; ++e) {
#pragma unroll
        for (int off = 16; off > 0; off >>= 1)
            acc[e] += __shfl_xor(acc[e], off, 32);
    }

    if (lane == 0) {
#pragma unroll
        for (int e = 0; e < NEXP; ++e) logits_out[t * NEXP + e] = acc[e];
        int i0 = 0; float v0 = acc[0];
#pragma unroll
        for (int e = 1; e < NEXP; ++e)
            if (acc[e] > v0) { v0 = acc[e]; i0 = e; }
        int i1 = -1; float v1 = -3.4e38f;
#pragma unroll
        for (int e = 0; e < NEXP; ++e)
            if (e != i0 && acc[e] > v1) { v1 = acc[e]; i1 = e; }
        float m  = fmaxf(v0, v1);
        float e0 = __expf(v0 - m), e1 = __expf(v1 - m);
        float s  = e0 + e1;
        topi[t * 2 + 0]  = i0;      topi[t * 2 + 1]  = i1;
        tgate[t * 2 + 0] = e0 / s;  tgate[t * 2 + 1] = e1 / s;
    }
}

// ---------------------------------------------------------------------------
// Scatter: per-expert (token*2+k, gate) lists via integer atomics.
// ---------------------------------------------------------------------------
__global__ __launch_bounds__(256) void scatter_kernel(
    const int* __restrict__ topi, const float* __restrict__ tgate,
    int* __restrict__ counts, int* __restrict__ tklist,
    float* __restrict__ glist) {
    const int t = blockIdx.x * blockDim.x + threadIdx.x;
    if (t >= T_TOK) return;
#pragma unroll
    for (int k = 0; k < TOPK; ++k) {
        int e = topi[t * 2 + k];
        int slot = atomicAdd(&counts[e], 1);
        tklist[e * T_TOK + slot] = t * 2 + k;
        glist[e * T_TOK + slot]  = tgate[t * 2 + k];
    }
}

// ---------------------------------------------------------------------------
// GEMM1 + SwiGLU:  h[tk, n0:n0+64] = silu(x.Win_a) * (x.Win_b)   (bf16 out)
// Block: 64 gathered tokens x 64 H-columns (a- and b-halves both computed).
// Async double-buffered LDS staging of w_in (already bf16).
// ---------------------------------------------------------------------------
__global__ __launch_bounds__(256) void gemm1_swiglu_kernel(
    const unsigned short* __restrict__ xbf,
    const unsigned short* __restrict__ winbf,
    const int* __restrict__ counts, const int* __restrict__ tklist,
    unsigned short* __restrict__ hbuf) {
    const int e  = blockIdx.z;
    const int m0 = blockIdx.y * 64;
    const int n0 = blockIdx.x * 64;              // column base in [0, H)
    const int cnt = counts[e];
    if (m0 >= cnt) return;

    __shared__ __align__(16) unsigned short sB[2][128 * LDSROW];
    __shared__ float sC[4][16][65];
    __shared__ int  sTK[64];

    const int tid = threadIdx.x;
    if (tid < 64) {
        int r = m0 + tid;
        sTK[tid] = (r < cnt) ? tklist[e * T_TOK + r] : -1;
    }
    __syncthreads();

    const int wid   = tid >> 5, lane = tid & 31;
    const int wm    = wid & 3,  half = wid >> 2;
    const int mrow  = lane & 15;
    const int khalf = lane >> 4;

    v8f cacc[4] = {};

    // staging map: each thread copies 64B (4 x b128) of one row per stage
    const int srow  = tid >> 1;                  // 0..127
    const int skoff = (tid & 1) * 32;            // bf16 units within row
    const int jrow  = (srow < 64) ? (n0 + srow) : (HID + n0 + (srow - 64));
    const unsigned short* bsrc =
        winbf + ((size_t)e * TWO_H + jrow) * DIM + skoff;
    unsigned short* sd0 = &sB[0][srow * LDSROW + skoff];
    unsigned short* sd1 = &sB[1][srow * LDSROW + skoff];

    // gather row; clamp padded rows to a valid one (row 0 exists since cnt>0) —
    // garbage accumulators for padded rows are masked at the store.
    int tkload = sTK[wm * 16 + mrow];
    if (tkload < 0) tkload = tklist[e * T_TOK];
    const unsigned short* asrc = xbf + (size_t)(tkload >> 1) * DIM + khalf * 8;

    // prologue: stage 0 into buffer 0
#pragma unroll
    for (int i = 0; i < 4; ++i) stage16(bsrc + i * 8, sd0 + i * 8);

    const int NST = DIM / KSTEP;
    for (int s = 0; s < NST; ++s) {
        wait_async();
        __syncthreads();   // buf[s&1] ready; buf[(s+1)&1] free for overwrite
        if (s + 1 < NST) {
            const unsigned short* src = bsrc + (s + 1) * KSTEP;
            unsigned short* dst = ((s + 1) & 1) ? sd1 : sd0;
#pragma unroll
            for (int i = 0; i < 4; ++i) stage16(src + i * 8, dst + i * 8);
        }
        const unsigned short* buf = sB[s & 1];
        const int kbase = s * KSTEP;
#pragma unroll
        for (int sub = 0; sub < 2; ++sub) {
            // A layout: K runs of 8 at khalf*8 and khalf*8+16 (unconditional)
            const unsigned short* ap = asrc + kbase + sub * 32;
            uint4 r0 = *(const uint4*)(ap);
            uint4 r1 = *(const uint4*)(ap + 16);
            Frag a;
            a.u[0] = r0.x; a.u[1] = r0.y; a.u[2] = r0.z; a.u[3] = r0.w;
            a.u[4] = r1.x; a.u[5] = r1.y; a.u[6] = r1.z; a.u[7] = r1.w;
#pragma unroll
            for (int j = 0; j < 4; ++j) {   // B: lane n = column, K contiguous
                const unsigned short* bp =
                    buf + (half * 64 + j * 16 + (lane & 15)) * LDSROW +
                    sub * 32 + khalf * 8;
                uint4 b0 = *(const uint4*)(bp);
                uint4 b1 = *(const uint4*)(bp + 16);
                Frag b;
                b.u[0] = b0.x; b.u[1] = b0.y; b.u[2] = b0.z; b.u[3] = b0.w;
                b.u[4] = b1.x; b.u[5] = b1.y; b.u[6] = b1.z; b.u[7] = b1.w;
                cacc[j] = __builtin_amdgcn_wmma_f32_16x16x32_bf16(
                    false, a.v, false, b.v, (short)0, cacc[j], false, false);
            }
        }
    }

    // b-half waves export their pre-activations through LDS; a-half applies SwiGLU
    const int nbase = lane & 15;
    const int mh    = (lane >> 4) * 8;
    if (half == 1) {
#pragma unroll
        for (int j = 0; j < 4; ++j)
#pragma unroll
            for (int r = 0; r < 8; ++r)
                sC[wm][mh + r][j * 16 + nbase] = cacc[j][r];
    }
    __syncthreads();
    if (half == 0) {
#pragma unroll
        for (int r = 0; r < 8; ++r) {
            int rowlocal = wm * 16 + mh + r;
            int tk = sTK[rowlocal];
            if (tk < 0) continue;
            unsigned short* hdst = hbuf + (size_t)tk * HID + n0;
#pragma unroll
            for (int j = 0; j < 4; ++j) {
                float av = cacc[j][r];
                float bv = sC[wm][mh + r][j * 16 + nbase];
                float hv = (av / (1.0f + __expf(-av))) * bv; // silu(a)*b
                hdst[j * 16 + nbase] = (unsigned short)f2bf(hv);
            }
        }
    }
}

// ---------------------------------------------------------------------------
// GEMM2: out[t, :] += gate * (h[tk, :] . w_out[e]^T).  Block: 64 rows x 128 D.
// ---------------------------------------------------------------------------
__global__ __launch_bounds__(256) void gemm2_kernel(
    const unsigned short* __restrict__ hbuf,
    const unsigned short* __restrict__ woutbf,
    const int* __restrict__ counts, const int* __restrict__ tklist,
    const float* __restrict__ glist, float* __restrict__ out) {
    const int e  = blockIdx.z;
    const int m0 = blockIdx.y * 64;
    const int n0 = blockIdx.x * 128;
    const int cnt = counts[e];
    if (m0 >= cnt) return;

    __shared__ __align__(16) unsigned short sB[2][128 * LDSROW];
    __shared__ int   sTK[64];
    __shared__ float sG[64];

    const int tid = threadIdx.x;
    if (tid < 64) {
        int r = m0 + tid;
        if (r < cnt) { sTK[tid] = tklist[e * T_TOK + r]; sG[tid] = glist[e * T_TOK + r]; }
        else         { sTK[tid] = -1;                    sG[tid] = 0.0f; }
    }
    __syncthreads();

    const int wid   = tid >> 5, lane = tid & 31;
    const int wm    = wid & 3,  wn   = wid >> 2;
    const int mrow  = lane & 15;
    const int khalf = lane >> 4;

    v8f cacc[4] = {};

    const int srow  = tid >> 1;
    const int skoff = (tid & 1) * 32;
    const unsigned short* bsrc =
        woutbf + ((size_t)e * DIM + (n0 + srow)) * HID + skoff;
    unsigned short* sd0 = &sB[0][srow * LDSROW + skoff];
    unsigned short* sd1 = &sB[1][srow * LDSROW + skoff];

    int tkload = sTK[wm * 16 + mrow];
    if (tkload < 0) tkload = tklist[e * T_TOK];
    const unsigned short* asrc = hbuf + (size_t)tkload * HID + khalf * 8;

#pragma unroll
    for (int i = 0; i < 4; ++i) stage16(bsrc + i * 8, sd0 + i * 8);

    const int NST = HID / KSTEP;
    for (int s = 0; s < NST; ++s) {
        wait_async();
        __syncthreads();
        if (s + 1 < NST) {
            const unsigned short* src = bsrc + (s + 1) * KSTEP;
            unsigned short* dst = ((s + 1) & 1) ? sd1 : sd0;
#pragma unroll
            for (int i = 0; i < 4; ++i) stage16(src + i * 8, dst + i * 8);
        }
        const unsigned short* buf = sB[s & 1];
        const int kbase = s * KSTEP;
#pragma unroll
        for (int sub = 0; sub < 2; ++sub) {
            const unsigned short* ap = asrc + kbase + sub * 32;
            uint4 r0 = *(const uint4*)(ap);
            uint4 r1 = *(const uint4*)(ap + 16);
            Frag a;
            a.u[0] = r0.x; a.u[1] = r0.y; a.u[2] = r0.z; a.u[3] = r0.w;
            a.u[4] = r1.x; a.u[5] = r1.y; a.u[6] = r1.z; a.u[7] = r1.w;
#pragma unroll
            for (int j = 0; j < 4; ++j) {
                const unsigned short* bp =
                    buf + (wn * 64 + j * 16 + (lane & 15)) * LDSROW +
                    sub * 32 + khalf * 8;
                uint4 b0 = *(const uint4*)(bp);
                uint4 b1 = *(const uint4*)(bp + 16);
                Frag b;
                b.u[0] = b0.x; b.u[1] = b0.y; b.u[2] = b0.z; b.u[3] = b0.w;
                b.u[4] = b1.x; b.u[5] = b1.y; b.u[6] = b1.z; b.u[7] = b1.w;
                cacc[j] = __builtin_amdgcn_wmma_f32_16x16x32_bf16(
                    false, a.v, false, b.v, (short)0, cacc[j], false, false);
            }
        }
    }

    const int nbase = lane & 15;
    const int mh    = (lane >> 4) * 8;
#pragma unroll
    for (int r = 0; r < 8; ++r) {
        int rowlocal = wm * 16 + mh + r;
        int tk = sTK[rowlocal];
        if (tk < 0) continue;
        float g   = sG[rowlocal];
        float* od = out + (size_t)(tk >> 1) * DIM + n0 + wn * 64;
#pragma unroll
        for (int j = 0; j < 4; ++j)
            atomicAdd(od + j * 16 + nbase, cacc[j][r] * g); // 2 adds/elem: order-invariant
    }
}

// ---------------------------------------------------------------------------
extern "C" void kernel_launch(void* const* d_in, const int* in_sizes, int n_in,
                              void* d_out, int out_size, void* d_ws, size_t ws_size,
                              hipStream_t stream) {
    (void)in_sizes; (void)n_in; (void)out_size; (void)ws_size;
    const float* x    = (const float*)d_in[0];  // [T, D]
    const float* wr   = (const float*)d_in[1];  // [E, D]
    const float* win  = (const float*)d_in[2];  // [E, 2H, D]
    const float* wout = (const float*)d_in[3];  // [E, D, H]

    float* out    = (float*)d_out;               // [T, D]
    float* logits = out + (size_t)T_TOK * DIM;   // [T, E]

    char* w = (char*)d_ws;
    int*   topi   = (int*)w;   w += (size_t)T_TOK * TOPK * sizeof(int);
    float* tgate  = (float*)w; w += (size_t)T_TOK * TOPK * sizeof(float);
    int*   counts = (int*)w;   w += 256;
    int*   tklist = (int*)w;   w += (size_t)NEXP * T_TOK * sizeof(int);
    float* glist  = (float*)w; w += (size_t)NEXP * T_TOK * sizeof(float);
    unsigned short* hbuf   = (unsigned short*)w; w += (size_t)T_TOK * TOPK * HID * 2;  // 8MB
    unsigned short* xbf    = (unsigned short*)w; w += (size_t)T_TOK * DIM * 2;         // 8MB
    unsigned short* winbf  = (unsigned short*)w; w += (size_t)NEXP * TWO_H * DIM * 2;  // 16MB
    unsigned short* woutbf = (unsigned short*)w;                                       // 8MB

    (void)hipMemsetAsync(d_out, 0, (size_t)T_TOK * DIM * sizeof(float), stream);
    (void)hipMemsetAsync(counts, 0, NEXP * sizeof(int), stream);

    // one-shot bf16 conversions
    cvt_bf16_kernel<<<(T_TOK * DIM / 8 + 255) / 256, 256, 0, stream>>>(
        x, xbf, T_TOK * DIM / 8);
    cvt_bf16_kernel<<<(NEXP * TWO_H * DIM / 8 + 255) / 256, 256, 0, stream>>>(
        win, winbf, NEXP * TWO_H * DIM / 8);
    cvt_bf16_kernel<<<(NEXP * DIM * HID / 8 + 255) / 256, 256, 0, stream>>>(
        wout, woutbf, NEXP * DIM * HID / 8);

    router_kernel<<<T_TOK / 8, 256, 0, stream>>>(x, wr, logits, topi, tgate);
    scatter_kernel<<<T_TOK / 256, 256, 0, stream>>>(topi, tgate, counts, tklist, glist);
    gemm1_swiglu_kernel<<<dim3(HID / 64, T_TOK / 64, NEXP), 256, 0, stream>>>(
        xbf, winbf, counts, tklist, hbuf);
    gemm2_kernel<<<dim3(DIM / 128, T_TOK / 64, NEXP), 256, 0, stream>>>(
        hbuf, woutbf, counts, tklist, glist, out);
}
